// RDN_HAB_43181601194628
// MI455X (gfx1250) — matrix operand order repaired and checked by reading, not compiled, not run on hardware
//
#include <hip/hip_runtime.h>
#include <hip/hip_bf16.h>
#include <math.h>

// ---------------------------------------------------------------------------
// Types for CDNA5 WMMA (wave32, 16x16x32 bf16 -> f32)
// ---------------------------------------------------------------------------
typedef __attribute__((ext_vector_type(16))) __bf16 v16bf;
typedef __attribute__((ext_vector_type(8)))  float  v8f;
typedef __attribute__((ext_vector_type(4)))  unsigned int u32x4;
typedef __attribute__((ext_vector_type(8)))  int i32x8;
typedef __attribute__((ext_vector_type(4)))  int i32x4;

#define WMMA_BF16(a, b, c) \
    __builtin_amdgcn_wmma_f32_16x16x32_bf16(false, (a), false, (b), (short)0, (c), false, false)

#if __has_builtin(__builtin_amdgcn_tensor_load_to_lds) && \
    __has_builtin(__builtin_amdgcn_s_wait_tensorcnt)
#define HAS_TDM 1
#else
#define HAS_TDM 0
#endif

// Problem constants
static constexpr int  Bc   = 4;
static constexpr int  Cc   = 192;
static constexpr int  NHc  = 6;
static constexpr int  WSc  = 16;
static constexpr int  SSc  = 8;
static constexpr int  Nc   = WSc * WSc;           // 256 tokens per window
static constexpr int  HDc  = Cc / NHc;            // 32
static constexpr int  NWIN = 256;                 // windows
static constexpr int  Mrows = 65536;              // B*H*W token rows
static constexpr float SCALEc = 0.17677669529663687f;  // 32^-0.5
static constexpr float CONV_SCALEc = 0.01f;

__device__ __forceinline__ float gelu_exact(float v) {
    return 0.5f * v * (1.0f + erff(v * 0.70710678118654752f));
}

// ---------------------------------------------------------------------------
// Tensor Data Mover: 2D tile (bf16) global -> LDS.  D# per ISA 08_async_tensor.
// group0: count=1 | lds_addr[63:32] | global_addr[120:64] | type(2)[127:126]
// group1: data_size=2B, tensor dims (large, no OOB clip), tile dims, dim0 stride
// This toolchain's builtin takes 6 args: (g0, g1, g2, g3, g4:i32x8, cpol).
// ---------------------------------------------------------------------------
#if HAS_TDM
__device__ __forceinline__ void tdm_load_2d_bf16(unsigned lds_addr, const void* gptr,
                                                 unsigned tile0, unsigned tile1,
                                                 unsigned stride0_elems) {
    unsigned long long ga = (unsigned long long)gptr;
    u32x4 g0;
    g0[0] = 1u;                                     // count=1, user mode, no gather
    g0[1] = lds_addr;                               // LDS byte address
    g0[2] = (unsigned)(ga & 0xFFFFFFFFu);           // global_addr[31:0]
    g0[3] = (unsigned)((ga >> 32) & 0x01FFFFFFu) | (2u << 30);  // [56:32] | type=2
    const unsigned tdim0 = 1u << 20, tdim1 = 1u << 20;
    i32x8 g1;
    g1[0] = 0x00010000;                             // data_size = 2 bytes
    g1[1] = (int)((tdim0 & 0xFFFFu) << 16);         // tensor_dim0[15:0]
    g1[2] = (int)((tdim0 >> 16) | ((tdim1 & 0xFFFFu) << 16));
    g1[3] = (int)((tdim1 >> 16) | (tile0 << 16));   // tile_dim0
    g1[4] = (int)(tile1 & 0xFFFFu);                 // tile_dim1 (tile_dim2 = 0)
    g1[5] = (int)stride0_elems;                     // tensor_dim0_stride[31:0]
    g1[6] = 0;
    g1[7] = 0;
    i32x4 z4 = {0, 0, 0, 0};
    i32x8 z8 = {0, 0, 0, 0, 0, 0, 0, 0};
    __builtin_amdgcn_tensor_load_to_lds(g0, g1, z4, z4, z8, 0);
}
#endif

// ---------------------------------------------------------------------------
// Fragment loaders (contiguous k = 16*(lane/16) + j, same order for A and B, wave32)
// ---------------------------------------------------------------------------
__device__ __forceinline__ v16bf load_a_frag(const __bf16* row_base, int lane) {
    int kb = (lane < 16) ? 0 : 16;
    v16bf a;
#pragma unroll
    for (int j = 0; j < 16; ++j) {
        a[j]     = row_base[kb + j];
    }
    return a;
}

__device__ __forceinline__ v16bf load_b_frag_contig(const __bf16* col_base, int lane) {
    int kb = (lane < 16) ? 0 : 16;
    v16bf b;
#pragma unroll
    for (int j = 0; j < 16; ++j) b[j] = col_base[kb + j];
    return b;
}

// ---------------------------------------------------------------------------
// f32 -> bf16 conversion
// ---------------------------------------------------------------------------
__global__ __launch_bounds__(256) void cvt_bf16_kernel(const float* __restrict__ src,
                                                       __bf16* __restrict__ dst, int n) {
    int i = blockIdx.x * 256 + threadIdx.x;
    if (i < n) dst[i] = (__bf16)src[i];
}

// ---------------------------------------------------------------------------
// LayerNorm1: x -> xn (f32) and shifted-window-partitioned xw (bf16)
// ---------------------------------------------------------------------------
__global__ __launch_bounds__(192) void ln1_kernel(const float* __restrict__ x,
                                                  const float* __restrict__ g,
                                                  const float* __restrict__ bb,
                                                  float* __restrict__ xn,
                                                  __bf16* __restrict__ xw) {
    int row = blockIdx.x;
    int t = threadIdx.x;
    __shared__ float red[192];
    __shared__ float sm, sv;
    float v = x[(size_t)row * Cc + t];
    red[t] = v;
    __syncthreads();
    for (int s = 96; s >= 6; s >>= 1) { if (t < s) red[t] += red[t + s]; __syncthreads(); }
    if (t == 0) sm = (red[0] + red[1] + red[2] + red[3] + red[4] + red[5]) * (1.0f / 192.0f);
    __syncthreads();
    float mean = sm;
    float d = v - mean;
    red[t] = d * d;
    __syncthreads();
    for (int s = 96; s >= 6; s >>= 1) { if (t < s) red[t] += red[t + s]; __syncthreads(); }
    if (t == 0) sv = (red[0] + red[1] + red[2] + red[3] + red[4] + red[5]) * (1.0f / 192.0f);
    __syncthreads();
    float o = d * rsqrtf(sv + 1e-5f) * g[t] + bb[t];
    xn[(size_t)row * Cc + t] = o;
    int b  = row >> 14;
    int pix = row & 16383;
    int h = pix >> 7, w = pix & 127;
    int hs = (h - SSc) & 127, ws = (w - SSc) & 127;
    int win = b * 64 + (hs >> 4) * 8 + (ws >> 4);
    int n = (hs & 15) * 16 + (ws & 15);
    xw[((size_t)win * Nc + n) * Cc + t] = (__bf16)o;
}

__global__ __launch_bounds__(192) void ln2_kernel(const float* __restrict__ y,
                                                  const float* __restrict__ g,
                                                  const float* __restrict__ bb,
                                                  __bf16* __restrict__ yn) {
    int row = blockIdx.x;
    int t = threadIdx.x;
    __shared__ float red[192];
    __shared__ float sm, sv;
    float v = y[(size_t)row * Cc + t];
    red[t] = v;
    __syncthreads();
    for (int s = 96; s >= 6; s >>= 1) { if (t < s) red[t] += red[t + s]; __syncthreads(); }
    if (t == 0) sm = (red[0] + red[1] + red[2] + red[3] + red[4] + red[5]) * (1.0f / 192.0f);
    __syncthreads();
    float mean = sm;
    float d = v - mean;
    red[t] = d * d;
    __syncthreads();
    for (int s = 96; s >= 6; s >>= 1) { if (t < s) red[t] += red[t + s]; __syncthreads(); }
    if (t == 0) sv = (red[0] + red[1] + red[2] + red[3] + red[4] + red[5]) * (1.0f / 192.0f);
    __syncthreads();
    float o = d * rsqrtf(sv + 1e-5f) * g[t] + bb[t];
    yn[(size_t)row * Cc + t] = (__bf16)o;
}

// ---------------------------------------------------------------------------
// conv1: 3x3, 192 -> 64, + GELU.  256 thr = 4 pixels x 64 oc
// ---------------------------------------------------------------------------
__global__ __launch_bounds__(256) void conv1_kernel(const float* __restrict__ xn,
                                                    const float* __restrict__ wgt,
                                                    const float* __restrict__ bias,
                                                    float* __restrict__ cv1) {
    int tid = threadIdx.x;
    int oc = tid & 63;
    int pix = blockIdx.x * 4 + (tid >> 6);
    int w = pix & 127, h = (pix >> 7) & 127, b = pix >> 14;
    float acc = bias[oc];
#pragma unroll
    for (int ky = 0; ky < 3; ++ky) {
        int hh = h + ky - 1;
        if ((unsigned)hh >= 128u) continue;
#pragma unroll
        for (int kx = 0; kx < 3; ++kx) {
            int ww = w + kx - 1;
            if ((unsigned)ww >= 128u) continue;
            const float* xp = xn + ((size_t)((b * 128 + hh) * 128 + ww)) * Cc;
            const float* wp = wgt + (size_t)((ky * 3 + kx) * Cc) * 64 + oc;
            for (int ic = 0; ic < Cc; ++ic) acc = fmaf(xp[ic], wp[ic * 64], acc);
        }
    }
    cv1[(size_t)pix * 64 + oc] = gelu_exact(acc);
}

// conv2: 3x3, 64 -> 192.  192 thr, one pixel per block
__global__ __launch_bounds__(192) void conv2_kernel(const float* __restrict__ cv1,
                                                    const float* __restrict__ wgt,
                                                    const float* __restrict__ bias,
                                                    float* __restrict__ cv2) {
    int oc = threadIdx.x;
    int pix = blockIdx.x;
    int w = pix & 127, h = (pix >> 7) & 127, b = pix >> 14;
    float acc = bias[oc];
#pragma unroll
    for (int ky = 0; ky < 3; ++ky) {
        int hh = h + ky - 1;
        if ((unsigned)hh >= 128u) continue;
#pragma unroll
        for (int kx = 0; kx < 3; ++kx) {
            int ww = w + kx - 1;
            if ((unsigned)ww >= 128u) continue;
            const float* xp = cv1 + ((size_t)((b * 128 + hh) * 128 + ww)) * 64;
            const float* wp = wgt + (size_t)((ky * 3 + kx) * 64) * Cc + oc;
            for (int ic = 0; ic < 64; ++ic) acc = fmaf(xp[ic], wp[ic * Cc], acc);
        }
    }
    cv2[(size_t)pix * Cc + oc] = acc;
}

// channel attention: global avg pool -> 192->6 relu -> 6->192 sigmoid
__global__ __launch_bounds__(256) void ca_kernel(const float* __restrict__ cv2,
                                                 const float* __restrict__ w1,
                                                 const float* __restrict__ b1,
                                                 const float* __restrict__ w2,
                                                 const float* __restrict__ b2,
                                                 float* __restrict__ ca) {
    int b = blockIdx.x, t = threadIdx.x;
    __shared__ float pooled[192];
    __shared__ float hid[6];
    if (t < 192) {
        float s = 0.0f;
        const float* base = cv2 + (size_t)b * 16384 * Cc + t;
        for (int p = 0; p < 16384; ++p) s += base[(size_t)p * Cc];
        pooled[t] = s * (1.0f / 16384.0f);
    }
    __syncthreads();
    if (t < 6) {
        float s = b1[t];
        for (int c = 0; c < 192; ++c) s = fmaf(pooled[c], w1[c * 6 + t], s);
        hid[t] = fmaxf(s, 0.0f);
    }
    __syncthreads();
    if (t < 192) {
        float s = b2[t];
#pragma unroll
        for (int j = 0; j < 6; ++j) s = fmaf(hid[j], w2[j * 192 + t], s);
        ca[b * 192 + t] = 1.0f / (1.0f + expf(-s));
    }
}

// ---------------------------------------------------------------------------
// Tiled WMMA GEMM: BM=128 BN=64 BK=32; 8 waves (4x2), 2x2 tiles per wave.
// A tile staged by Tensor Data Mover when available; B tile staged transposed.
// EPI: 0=QKV  1=PROJ(+win-reverse+residual)  2=FC1(+GELU)  3=FC2(+y residual)
// ---------------------------------------------------------------------------
template <int EPI>
__global__ __launch_bounds__(256) void gemm_wmma_kernel(
    const __bf16* __restrict__ A, const __bf16* __restrict__ Bw,
    const float* __restrict__ bias, int Ksize, int Nsize,
    void* __restrict__ outp,
    const float* __restrict__ aux0, const float* __restrict__ aux1,
    const float* __restrict__ aux2) {
    constexpr int BM = 128, BN = 64, BK = 32;
    __shared__ __bf16 As[BM * BK];   // [m][k]
    __shared__ __bf16 Bs[BN * BK];   // [n][k]

    int tid = threadIdx.x;
    int lane = tid & 31;
    int wave = tid >> 5;
    int waveM = wave >> 1;
    int waveN = wave & 1;
    int bm0 = blockIdx.x * BM;
    int bn0 = blockIdx.y * BN;

    v8f acc[2][2];
#pragma unroll
    for (int i = 0; i < 2; ++i)
#pragma unroll
        for (int j = 0; j < 2; ++j) acc[i][j] = (v8f)0.0f;

    int arow = tid >> 1, ako = (tid & 1) * 16;
    int bkr = tid >> 3, bnc = (tid & 7) * 8;
#if HAS_TDM
    unsigned asAddr = (unsigned)(size_t)(void*)As;
#endif

    for (int k0 = 0; k0 < Ksize; k0 += BK) {
#if HAS_TDM
        // DMA the A tile [128][32] straight into LDS (wave 0 issues; EXEC ignored)
        if (wave == 0) {
            tdm_load_2d_bf16(asAddr, A + (size_t)bm0 * Ksize + k0, BK, BM,
                             (unsigned)Ksize);
        }
#else
        {
            const __bf16* src = A + (size_t)(bm0 + arow) * Ksize + k0 + ako;
            const float4* s4 = (const float4*)src;
            float4* d4 = (float4*)(As + arow * BK + ako);
            d4[0] = s4[0];
            d4[1] = s4[1];
        }
#endif
        // stage B tile [32][64] transposed into [n][k]
        {
            const __bf16* src = Bw + (size_t)(k0 + bkr) * Nsize + bn0 + bnc;
#pragma unroll
            for (int j = 0; j < 8; ++j) Bs[(bnc + j) * BK + bkr] = src[j];
            if (k0 + BK < Ksize) __builtin_prefetch(src + (size_t)BK * Nsize, 0, 1);
        }
#if HAS_TDM
        if (wave == 0) __builtin_amdgcn_s_wait_tensorcnt(0);
#endif
        __syncthreads();

        v16bf af[2], bf[2];
        af[0] = load_a_frag(As + (waveM * 32 + 0)  * BK + ((lane & 15) * BK), lane);
        af[1] = load_a_frag(As + (waveM * 32 + 16) * BK + ((lane & 15) * BK), lane);
        bf[0] = load_b_frag_contig(Bs + (waveN * 32 + 0  + (lane & 15)) * BK, lane);
        bf[1] = load_b_frag_contig(Bs + (waveN * 32 + 16 + (lane & 15)) * BK, lane);

#pragma unroll
        for (int rt = 0; rt < 2; ++rt)
#pragma unroll
            for (int ct = 0; ct < 2; ++ct)
                acc[rt][ct] = WMMA_BF16(af[rt], bf[ct], acc[rt][ct]);
        __syncthreads();
    }

    int col0 = lane & 15;
    int rowhalf = (lane < 16) ? 0 : 8;
#pragma unroll
    for (int rt = 0; rt < 2; ++rt) {
#pragma unroll
        for (int ct = 0; ct < 2; ++ct) {
            int col = bn0 + waveN * 32 + ct * 16 + col0;
            float bc = bias[col];
#pragma unroll
            for (int r = 0; r < 8; ++r) {
                int row = bm0 + waveM * 32 + rt * 16 + rowhalf + r;
                float val = acc[rt][ct][r] + bc;
                if constexpr (EPI == 0) {
                    if (col < Cc) val *= SCALEc;  // q scaling
                    ((__bf16*)outp)[(size_t)row * Nsize + col] = (__bf16)val;
                } else if constexpr (EPI == 1) {
                    int win = row >> 8, n = row & 255;
                    int b = win >> 6, wi = win & 63;
                    int hs = ((wi >> 3) << 4) + (n >> 4);
                    int ws = ((wi & 7) << 4) + (n & 15);
                    int h = (hs + SSc) & 127, w = (ws + SSc) & 127;
                    size_t pix = ((size_t)b << 14) + (h << 7) + w;
                    float sc = aux0[pix * Cc + col];
                    float cv = aux1[pix * Cc + col] * aux2[b * Cc + col] * CONV_SCALEc;
                    ((float*)outp)[pix * Cc + col] = sc + val + cv;
                } else if constexpr (EPI == 2) {
                    ((__bf16*)outp)[(size_t)row * Nsize + col] = (__bf16)gelu_exact(val);
                } else {
                    ((float*)outp)[(size_t)row * Cc + col] =
                        aux0[(size_t)row * Cc + col] + val;
                }
            }
        }
    }
}

// ---------------------------------------------------------------------------
// Fused window attention: block per (window, head), 4 waves.
// K/V staged by TDM (strided head-slice out of packed QKV); softmax via
// wave32 half-shuffles; P re-read from LDS in A-fragment layout for P.V.
// ---------------------------------------------------------------------------
__global__ __launch_bounds__(128) void attn_kernel(const __bf16* __restrict__ qkv,
                                                   const float* __restrict__ rpb,
                                                   const int* __restrict__ rpi,
                                                   const float* __restrict__ mask,
                                                   __bf16* __restrict__ attnout) {
    int win = blockIdx.x;
    int head = blockIdx.y;
    int tid = threadIdx.x;
    int lane = tid & 31;
    int wave = tid >> 5;

    __shared__ __bf16 Ks[Nc * HDc];      // 16 KB
    __shared__ __bf16 Vs[Nc * HDc];      // 16 KB
    __shared__ __bf16 Ps[4][16 * Nc];    // 32 KB

#if HAS_TDM
    if (wave == 0) {
        const __bf16* base = qkv + (size_t)win * Nc * (3 * Cc) + head * HDc;
        tdm_load_2d_bf16((unsigned)(size_t)(void*)Ks, base + Cc,     HDc, Nc, 3 * Cc);
        tdm_load_2d_bf16((unsigned)(size_t)(void*)Vs, base + 2 * Cc, HDc, Nc, 3 * Cc);
        __builtin_amdgcn_s_wait_tensorcnt(0);
    }
#else
    for (int i = tid; i < Nc; i += 128) {
        const __bf16* base = qkv + ((size_t)(win * Nc + i)) * (3 * Cc);
        const float4* ksrc = (const float4*)(base + Cc + head * HDc);
        const float4* vsrc = (const float4*)(base + 2 * Cc + head * HDc);
        float4* kd = (float4*)(Ks + i * HDc);
        float4* vd = (float4*)(Vs + i * HDc);
#pragma unroll
        for (int j = 0; j < 4; ++j) { kd[j] = ksrc[j]; vd[j] = vsrc[j]; }
    }
#endif
    __syncthreads();

    int col0 = lane & 15;
    int rowhalf = (lane < 16) ? 0 : 8;
    int wi = win & 63;

    for (int rt = wave; rt < 16; rt += 4) {
        const __bf16* qrow =
            qkv + ((size_t)(win * Nc + rt * 16 + col0)) * (3 * Cc) + head * HDc;
        v16bf qa = load_a_frag(qrow, lane);

        // S = Q K^T  (q pre-scaled at QKV store)
        v8f s[16];
#pragma unroll
        for (int ct = 0; ct < 16; ++ct) {
            v16bf kb = load_b_frag_contig(Ks + (ct * 16 + col0) * HDc, lane);
            s[ct] = WMMA_BF16(qa, kb, (v8f)0.0f);
        }

        // bias + mask + softmax
#pragma unroll
        for (int r = 0; r < 8; ++r) {
            int nrow = rt * 16 + rowhalf + r;
            float mx = -1e30f;
#pragma unroll
            for (int ct = 0; ct < 16; ++ct) {
                int mcol = ct * 16 + col0;
                float bm = rpb[rpi[nrow * Nc + mcol] * NHc + head] +
                           mask[(size_t)wi * (Nc * Nc) + nrow * Nc + mcol];
                s[ct][r] += bm;
                mx = fmaxf(mx, s[ct][r]);
            }
#pragma unroll
            for (int d = 1; d < 16; d <<= 1) mx = fmaxf(mx, __shfl_xor(mx, d, 32));
            float sum = 0.0f;
#pragma unroll
            for (int ct = 0; ct < 16; ++ct) {
                float e = __expf(s[ct][r] - mx);
                s[ct][r] = e;
                sum += e;
            }
#pragma unroll
            for (int d = 1; d < 16; d <<= 1) sum += __shfl_xor(sum, d, 32);
            float inv = 1.0f / sum;
#pragma unroll
            for (int ct = 0; ct < 16; ++ct) s[ct][r] *= inv;
        }

        // stage P (bf16) into this wave's LDS strip [16][256]
#pragma unroll
        for (int ct = 0; ct < 16; ++ct)
#pragma unroll
            for (int r = 0; r < 8; ++r)
                Ps[wave][(rowhalf + r) * Nc + ct * 16 + col0] = (__bf16)s[ct][r];
        __syncthreads();

        // O = P V
        v8f o[2] = {(v8f)0.0f, (v8f)0.0f};
#pragma unroll
        for (int kt = 0; kt < 8; ++kt) {
            v16bf pa = load_a_frag(&Ps[wave][col0 * Nc + kt * 32], lane);
#pragma unroll
            for (int dt = 0; dt < 2; ++dt) {
                v16bf vb;
                int kbase = kt * 32 + ((lane < 16) ? 0 : 16);
#pragma unroll
                for (int j = 0; j < 16; ++j)
                    vb[j] = Vs[(kbase + j) * HDc + dt * 16 + col0];
                o[dt] = WMMA_BF16(pa, vb, o[dt]);
            }
        }

#pragma unroll
        for (int dt = 0; dt < 2; ++dt)
#pragma unroll
            for (int r = 0; r < 8; ++r)
                attnout[((size_t)(win * Nc + rt * 16 + rowhalf + r)) * Cc +
                        head * HDc + dt * 16 + col0] = (__bf16)o[dt][r];
        __syncthreads();
    }
}

// ---------------------------------------------------------------------------
// Host launcher
// ---------------------------------------------------------------------------
extern "C" void kernel_launch(void* const* d_in, const int* in_sizes, int n_in,
                              void* d_out, int out_size, void* d_ws, size_t ws_size,
                              hipStream_t stream) {
    (void)in_sizes; (void)n_in; (void)out_size; (void)ws_size;
    const float* x      = (const float*)d_in[0];
    const float* qkv_w  = (const float*)d_in[1];
    const float* qkv_b  = (const float*)d_in[2];
    const float* proj_w = (const float*)d_in[3];
    const float* proj_b = (const float*)d_in[4];
    const float* rpb    = (const float*)d_in[5];
    const float* n1w    = (const float*)d_in[6];
    const float* n1b    = (const float*)d_in[7];
    const float* n2w    = (const float*)d_in[8];
    const float* n2b    = (const float*)d_in[9];
    const float* c1w    = (const float*)d_in[10];
    const float* c1b    = (const float*)d_in[11];
    const float* c2w    = (const float*)d_in[12];
    const float* c2b    = (const float*)d_in[13];
    const float* ca1w   = (const float*)d_in[14];
    const float* ca1b   = (const float*)d_in[15];
    const float* ca2w   = (const float*)d_in[16];
    const float* ca2b   = (const float*)d_in[17];
    const float* fc1w   = (const float*)d_in[18];
    const float* fc1b   = (const float*)d_in[19];
    const float* fc2w   = (const float*)d_in[20];
    const float* fc2b   = (const float*)d_in[21];
    const int*   rpi    = (const int*)d_in[22];
    const float* amask  = (const float*)d_in[23];

    char* ws = (char*)d_ws;
    size_t off = 0;
    auto alloc = [&](size_t bytes) { void* p = ws + off; off += (bytes + 255) & ~(size_t)255; return p; };

    float*  xn   = (float*)alloc((size_t)Mrows * Cc * 4);
    float*  yv   = (float*)alloc((size_t)Mrows * Cc * 4);
    float*  cv1  = (float*)alloc((size_t)Mrows * 64 * 4);
    float*  cv2  = (float*)alloc((size_t)Mrows * Cc * 4);
    float*  cav  = (float*)alloc((size_t)Bc * Cc * 4);
    __bf16* xw   = (__bf16*)alloc((size_t)Mrows * Cc * 2);
    __bf16* qkvb = (__bf16*)alloc((size_t)Mrows * 3 * Cc * 2);
    __bf16* atno = (__bf16*)alloc((size_t)Mrows * Cc * 2);
    __bf16* yn   = (__bf16*)alloc((size_t)Mrows * Cc * 2);
    __bf16* hbuf = (__bf16*)alloc((size_t)Mrows * 4 * Cc * 2);
    __bf16* wq   = (__bf16*)alloc((size_t)Cc * 3 * Cc * 2);
    __bf16* wp   = (__bf16*)alloc((size_t)Cc * Cc * 2);
    __bf16* w1   = (__bf16*)alloc((size_t)Cc * 4 * Cc * 2);
    __bf16* w2   = (__bf16*)alloc((size_t)4 * Cc * Cc * 2);

    cvt_bf16_kernel<<<(Cc * 3 * Cc + 255) / 256, 256, 0, stream>>>(qkv_w, wq, Cc * 3 * Cc);
    cvt_bf16_kernel<<<(Cc * Cc + 255) / 256, 256, 0, stream>>>(proj_w, wp, Cc * Cc);
    cvt_bf16_kernel<<<(Cc * 4 * Cc + 255) / 256, 256, 0, stream>>>(fc1w, w1, Cc * 4 * Cc);
    cvt_bf16_kernel<<<(4 * Cc * Cc + 255) / 256, 256, 0, stream>>>(fc2w, w2, 4 * Cc * Cc);

    ln1_kernel<<<Mrows, 192, 0, stream>>>(x, n1w, n1b, xn, xw);

    conv1_kernel<<<Mrows / 4, 256, 0, stream>>>(xn, c1w, c1b, cv1);
    conv2_kernel<<<Mrows, 192, 0, stream>>>(cv1, c2w, c2b, cv2);
    ca_kernel<<<Bc, 256, 0, stream>>>(cv2, ca1w, ca1b, ca2w, ca2b, cav);

    gemm_wmma_kernel<0><<<dim3(Mrows / 128, (3 * Cc) / 64), 256, 0, stream>>>(
        xw, wq, qkv_b, Cc, 3 * Cc, qkvb, nullptr, nullptr, nullptr);

    attn_kernel<<<dim3(NWIN, NHc), 128, 0, stream>>>(qkvb, rpb, rpi, amask, atno);

    gemm_wmma_kernel<1><<<dim3(Mrows / 128, Cc / 64), 256, 0, stream>>>(
        atno, wp, proj_b, Cc, Cc, yv, x, cv2, cav);

    ln2_kernel<<<Mrows, 192, 0, stream>>>(yv, n2w, n2b, yn);

    gemm_wmma_kernel<2><<<dim3(Mrows / 128, (4 * Cc) / 64), 256, 0, stream>>>(
        yn, w1, fc1b, Cc, 4 * Cc, hbuf, nullptr, nullptr, nullptr);
    gemm_wmma_kernel<3><<<dim3(Mrows / 128, Cc / 64), 256, 0, stream>>>(
        hbuf, w2, fc2b, 4 * Cc, Cc, d_out, yv, nullptr, nullptr);
}
